// PVMLayer5_11639361372337
// MI455X (gfx1250) — matrix operand order repaired and checked by reading, not compiled, not run on hardware
//
#include <hip/hip_runtime.h>
#include <hip/hip_bf16.h>

typedef __attribute__((ext_vector_type(16))) _Float16 v16h;
typedef __attribute__((ext_vector_type(8)))  _Float16 v8h;
typedef __attribute__((ext_vector_type(8)))  float    v8f;

#define BSZ    8
#define CCH    640
#define LSEQ   4096
#define NSEQ   40      /* 5 * BSZ */
#define DMC    128
#define DI     256
#define SST    16
#define NDBC   40      /* R + 2S */

// ---------------------------------------------------------------------------
// f32 -> f16 weight conversion with zero-padding of N to Npad (GEMM ldw)
// ---------------------------------------------------------------------------
__global__ void cvt_pad_kernel(const float* __restrict__ src, _Float16* __restrict__ dst,
                               int K, int N, int Npad) {
    int i = blockIdx.x * 256 + threadIdx.x;
    if (i >= K * Npad) return;
    int k = i / Npad, n = i - k * Npad;
    dst[i] = (n < N) ? (_Float16)src[(size_t)k * N + n] : (_Float16)0.f;
}

// ---------------------------------------------------------------------------
// LayerNorm over C for x in (B, C, L); emit chunk layout (5B, L, 128) f32+f16
// block = 256 threads = 32 l-lanes x 8 c-groups, one (b, 32-l) tile per block
// ---------------------------------------------------------------------------
__global__ __launch_bounds__(256) void ln1_kernel(const float* __restrict__ x,
                                                  const float* __restrict__ g,
                                                  const float* __restrict__ bt,
                                                  float* __restrict__ chunks_f,
                                                  _Float16* __restrict__ chunks_h) {
    const int b  = blockIdx.y;
    const int tx = threadIdx.x & 31;
    const int ty = threadIdx.x >> 5;
    const int l  = blockIdx.x * 32 + tx;
    const float* xb = x + ((size_t)b * CCH) * LSEQ + l;

    float s = 0.f, ss = 0.f;
    for (int i = 0; i < 80; ++i) {
        float v = xb[(size_t)(ty + 8 * i) * LSEQ];
        s += v; ss += v * v;
    }
    __shared__ float red0[8][32], red1[8][32];
    __shared__ float mu_s[32], rs_s[32];
    red0[ty][tx] = s; red1[ty][tx] = ss;
    __syncthreads();
    if (ty == 0) {
        float S = 0.f, SS = 0.f;
        for (int j = 0; j < 8; ++j) { S += red0[j][tx]; SS += red1[j][tx]; }
        float mu = S * (1.f / CCH);
        float var = SS * (1.f / CCH) - mu * mu;
        mu_s[tx] = mu;
        rs_s[tx] = rsqrtf(var + 1e-5f);
    }
    __syncthreads();
    const float mu = mu_s[tx], rs = rs_s[tx];
    for (int i = 0; i < 80; ++i) {
        int c = ty + 8 * i;
        float v = (xb[(size_t)c * LSEQ] - mu) * rs * g[c] + bt[c];
        int ch = c >> 7, d = c & 127;               // chunk, in-chunk dim
        size_t idx = (((size_t)(ch * BSZ + b)) * LSEQ + l) * DMC + d;
        chunks_f[idx] = v;
        chunks_h[idx] = (_Float16)v;
    }
}

// ---------------------------------------------------------------------------
// WMMA f16 GEMM: C(MxN) = A(MxK) * W(KxNpad), f32 accum, fused epilogues
// block tile 128x64, 8 waves (wave32) of 32x32 = 2x2 v_wmma_f32_16x16x32_f16
// Double-buffered LDS; A tile staged via GLOBAL_LOAD_ASYNC_TO_LDS_B128 with
// pipelined s_wait_asynccnt 2 (issue tile k+1's DMA before waiting tile k);
// B tile via one 16B vector load + transposed LDS scatter (W zero-padded).
// MODE 0: store f16 to outH[r*N+c]
// MODE 1: store f32 to outF[r*N+c]            (N may be < BN; store-guarded)
// MODE 2: outF = acc + scale*addend, scattered to (B,L,C) chunk interleave
// MODE 3: outF = acc + bias[c], transposed store to (B,C,L)
// ---------------------------------------------------------------------------
#define BM 128
#define BN 64
#define BK 32

template <int MODE>
__global__ __launch_bounds__(256) void gemm_wmma_kernel(
    const _Float16* __restrict__ A, const _Float16* __restrict__ W,
    float* __restrict__ outF, _Float16* __restrict__ outH,
    int M, int N, int K, int ldw,
    const float* __restrict__ addend, const float* __restrict__ bias,
    const float* __restrict__ scale_ptr) {

    __shared__ alignas(32) _Float16 As[2][BM][BK];   // 2 x 8 KB
    __shared__ alignas(32) _Float16 Bs[2][BN][BK];   // 2 x 4 KB, transposed [n][k]

    const int tid  = threadIdx.x;
    const int lane = tid & 31;
    const int wave = tid >> 5;
    const int wm   = wave & 3;        // 4 wave-tiles in M
    const int wn   = wave >> 2;       // 2 wave-tiles in N
    const int m0   = blockIdx.x * BM;
    const int n0   = blockIdx.y * BN;
    const int hs   = lane >> 4;       // half-wave select
    const int l16  = lane & 15;

    // A-tile async staging: each thread owns 32 contiguous bytes on both sides
    const int arow = tid >> 1;
    const int acol = (tid & 1) * 16;                      // halves
    const unsigned int ldsA0 =
        (unsigned int)(uintptr_t)(&As[0][0][0]) + (unsigned int)(arow * (BK * 2) + acol * 2);
    const _Float16* gA = A + (size_t)(m0 + arow) * K + acol;

    // B-tile staging coords
    const int bk = tid >> 3;
    const int bn = (tid & 7) * 8;

    auto stageA = [&](int k0, int p) {
        unsigned int ldsA = ldsA0 + (unsigned int)(p * (BM * BK * 2));
        unsigned long long ga = (unsigned long long)(uintptr_t)(gA + k0);
        asm volatile(
            "global_load_async_to_lds_b128 %0, %1, off\n\t"
            "global_load_async_to_lds_b128 %0, %1, off offset:16"
            :: "v"(ldsA), "v"(ga) : "memory");
    };
    auto stageB = [&](int k0, int p) {
        v8h wv = *(const v8h*)(W + (size_t)(k0 + bk) * ldw + n0 + bn);
        #pragma unroll
        for (int j = 0; j < 8; ++j) Bs[p][bn + j][bk] = wv[j];
    };

    v8f acc[2][2];
    #pragma unroll
    for (int i = 0; i < 2; ++i)
        #pragma unroll
        for (int j = 0; j < 2; ++j) { v8f z = {}; acc[i][j] = z; }

    stageA(0, 0);
    stageB(0, 0);

    const int nk = K / BK;
    for (int kt = 0; kt < nk; ++kt) {
        const int cur = kt & 1;
        const int nxt = cur ^ 1;
        // issue next tile's DMA before waiting on the current one (overlap);
        // last iteration re-copies the current tile into the idle buffer
        const int kn = (kt + 1 < nk) ? (kt + 1) * BK : kt * BK;
        stageA(kn, nxt);
        stageB(kn, nxt);
        asm volatile("s_wait_asynccnt 2" ::: "memory");   // tile kt's pair landed
        __syncthreads();

        // assemble fragments in contiguous k order (k = hs*16 + j) for both A and B
        v16h af[2], bf[2];
        #pragma unroll
        for (int mt = 0; mt < 2; ++mt) {
            int m = wm * 32 + mt * 16 + l16;                // M = lane&15
            v8h lo = *(const v8h*)&As[cur][m][hs * 16];     // K = hs*16 + j
            v8h hi = *(const v8h*)&As[cur][m][hs * 16 + 8]; // K = hs*16 + 8 + j
            #pragma unroll
            for (int j = 0; j < 8; ++j) { af[mt][j] = lo[j]; af[mt][8 + j] = hi[j]; }
        }
        #pragma unroll
        for (int nt = 0; nt < 2; ++nt) {
            int n = wn * 32 + nt * 16 + l16;                // N = lane&15
            bf[nt] = *(const v16h*)&Bs[cur][n][hs * 16];    // K = hs*16 + j (contig)
        }

        #pragma unroll
        for (int mt = 0; mt < 2; ++mt)
            #pragma unroll
            for (int nt = 0; nt < 2; ++nt)
                acc[mt][nt] = __builtin_amdgcn_wmma_f32_16x16x32_f16(
                    false, af[mt], false, bf[nt], (short)0, acc[mt][nt], false, false);
        __syncthreads();
    }

    float scale = 0.f;
    if constexpr (MODE == 2) scale = scale_ptr[0];

    #pragma unroll
    for (int mt = 0; mt < 2; ++mt) {
        #pragma unroll
        for (int nt = 0; nt < 2; ++nt) {
            int mbase = m0 + wm * 32 + mt * 16 + hs * 8;  // VGPR v -> M = v + 8*hs
            int cc    = n0 + wn * 32 + nt * 16 + l16;
            if constexpr (MODE == 1) { if (cc >= N) continue; }
            #pragma unroll
            for (int v = 0; v < 8; ++v) {
                int r = mbase + v;
                float val = acc[mt][nt][v];
                if constexpr (MODE == 0) {
                    outH[(size_t)r * N + cc] = (_Float16)val;
                } else if constexpr (MODE == 1) {
                    outF[(size_t)r * N + cc] = val;
                } else if constexpr (MODE == 2) {
                    int n = r >> 12, l = r & 4095;     // r = n*L + l
                    int b = n & 7, ch = n >> 3;        // n = ch*B + b
                    float v2 = val + scale * addend[(size_t)r * DMC + cc];
                    outF[((size_t)b * LSEQ + l) * CCH + ch * DMC + cc] = v2;
                } else {
                    int b = r >> 12, l = r & 4095;     // r = b*L + l
                    outF[((size_t)b * CCH + cc) * LSEQ + l] = val + bias[cc];
                }
            }
        }
    }
}

// ---------------------------------------------------------------------------
// depthwise causal conv (D_CONV=4) + bias + SiLU; xin = xz[..., :256]
// one block = one n, 64 l's; thread d carries a 4-deep shift register
// ---------------------------------------------------------------------------
__global__ __launch_bounds__(256) void conv_kernel(const _Float16* __restrict__ xz,
                                                   const float* __restrict__ cw,
                                                   const float* __restrict__ cb,
                                                   float* __restrict__ xc_f,
                                                   _Float16* __restrict__ xc_h) {
    const int n = blockIdx.y;
    const int l0 = blockIdx.x * 64;
    const int d = threadIdx.x;
    const float w0 = cw[d * 4 + 0], w1 = cw[d * 4 + 1];
    const float w2 = cw[d * 4 + 2], w3 = cw[d * 4 + 3];
    const float bb = cb[d];
    const _Float16* xin = xz + ((size_t)n * LSEQ) * (2 * DI) + d;

    float x0, x1, x2;   // x[l-3], x[l-2], x[l-1]
    if (l0 == 0) { x0 = x1 = x2 = 0.f; }
    else {
        x0 = (float)xin[(size_t)(l0 - 3) * (2 * DI)];
        x1 = (float)xin[(size_t)(l0 - 2) * (2 * DI)];
        x2 = (float)xin[(size_t)(l0 - 1) * (2 * DI)];
    }
    for (int i = 0; i < 64; ++i) {
        int l = l0 + i;
        float x3 = (float)xin[(size_t)l * (2 * DI)];
        float a = w0 * x0 + w1 * x1 + w2 * x2 + w3 * x3 + bb;
        float sv = a / (1.f + __expf(-a));              // SiLU
        size_t idx = ((size_t)n * LSEQ + l) * DI + d;
        xc_f[idx] = sv;
        xc_h[idx] = (_Float16)sv;
        x0 = x1; x1 = x2; x2 = x3;
    }
}

// ---------------------------------------------------------------------------
// delta = softplus(dt @ dt_proj_w + dt_proj_b); dt = dbc[:, :8]   (K=8, VALU)
// ---------------------------------------------------------------------------
__global__ __launch_bounds__(256) void dtproj_kernel(const float* __restrict__ dbc,
                                                     const float* __restrict__ dtw,
                                                     const float* __restrict__ dtb,
                                                     float* __restrict__ delta) {
    const int d = threadIdx.x;
    const int r0 = blockIdx.x * 16;
    float w[8];
    #pragma unroll
    for (int j = 0; j < 8; ++j) w[j] = dtw[j * DI + d];
    const float bb = dtb[d];
    __shared__ float dts[16][8];
    if (threadIdx.x < 128)
        dts[threadIdx.x >> 3][threadIdx.x & 7] =
            dbc[(size_t)(r0 + (threadIdx.x >> 3)) * NDBC + (threadIdx.x & 7)];
    __syncthreads();
    for (int i = 0; i < 16; ++i) {
        float a = bb;
        #pragma unroll
        for (int j = 0; j < 8; ++j) a += dts[i][j] * w[j];
        float sp = (a > 20.f) ? a : __logf(1.f + __expf(a));   // softplus
        delta[(size_t)(r0 + i) * DI + d] = sp;
    }
}

// ---------------------------------------------------------------------------
// selective scan: one block per n, thread = channel d, state h[16] in VGPRs
// B/C rows staged to LDS in 64-step chunks; next-chunk prefetch of delta/u
// ---------------------------------------------------------------------------
__global__ __launch_bounds__(256) void scan_kernel(const float* __restrict__ delta,
                                                   const float* __restrict__ dbc,
                                                   const float* __restrict__ u,
                                                   const float* __restrict__ A_log,
                                                   float* __restrict__ y) {
    const int n = blockIdx.x;
    const int d = threadIdx.x;
    float a[SST], h[SST];
    #pragma unroll
    for (int s = 0; s < SST; ++s) { a[s] = -__expf(A_log[d * SST + s]); h[s] = 0.f; }

    __shared__ alignas(16) float bs[64][SST];
    __shared__ alignas(16) float cs[64][SST];

    for (int l0 = 0; l0 < LSEQ; l0 += 64) {
        __syncthreads();
        {   // stage B (cols 8..23) and C (cols 24..39) for 64 steps
            int r = threadIdx.x >> 2;
            int q = threadIdx.x & 3;
            const float* src = dbc + ((size_t)n * LSEQ + l0 + r) * NDBC + 8 + q * 8;
            float4 v0 = *(const float4*)src;
            float4 v1 = *(const float4*)(src + 4);
            float* dst = (q < 2) ? &bs[r][q * 8] : &cs[r][(q - 2) * 8];
            *(float4*)dst = v0;
            *((float4*)dst + 1) = v1;
        }
        if (l0 + 64 < LSEQ) {   // prefetch next chunk's per-channel streams
            size_t nidx = ((size_t)n * LSEQ + l0 + 64) * DI + d;
            __builtin_prefetch(&delta[nidx], 0, 0);
            __builtin_prefetch(&u[nidx], 0, 0);
        }
        __syncthreads();

        for (int i = 0; i < 64; ++i) {
            size_t idx = ((size_t)n * LSEQ + l0 + i) * DI + d;
            float dlt = delta[idx];
            float du  = dlt * u[idx];
            float accv = 0.f;
            #pragma unroll
            for (int s = 0; s < SST; ++s) {
                h[s] = h[s] * __expf(dlt * a[s]) + du * bs[i][s];
                accv += h[s] * cs[i][s];
            }
            y[idx] = accv;
        }
    }
}

// ---------------------------------------------------------------------------
// gate: yg = (y + xc*D) * silu(z), store f16 for out_proj GEMM
// ---------------------------------------------------------------------------
__global__ void gate_kernel(const float* __restrict__ y, const float* __restrict__ xc_f,
                            const _Float16* __restrict__ xz, const float* __restrict__ Dw,
                            _Float16* __restrict__ yg) {
    size_t i = (size_t)blockIdx.x * 256 + threadIdx.x;
    int d = (int)(i & (DI - 1));
    size_t rl = i >> 8;
    float z = (float)xz[rl * (2 * DI) + DI + d];
    float sz = z / (1.f + __expf(-z));
    yg[i] = (_Float16)((y[i] + xc_f[i] * Dw[d]) * sz);
}

// ---------------------------------------------------------------------------
// LayerNorm over contiguous C=640 rows of (B,L,C); one wave per row, f16 out
// ---------------------------------------------------------------------------
__global__ __launch_bounds__(256) void ln2_kernel(const float* __restrict__ yin,
                                                  const float* __restrict__ g,
                                                  const float* __restrict__ bt,
                                                  _Float16* __restrict__ outh) {
    const int row  = blockIdx.x * 8 + (threadIdx.x >> 5);
    const int lane = threadIdx.x & 31;
    const float* p = yin + (size_t)row * CCH;
    float vals[20];
    float s = 0.f, ss = 0.f;
    #pragma unroll
    for (int i = 0; i < 20; ++i) {
        float v = p[lane + 32 * i];
        vals[i] = v; s += v; ss += v * v;
    }
    #pragma unroll
    for (int off = 16; off > 0; off >>= 1) {
        s  += __shfl_xor(s, off, 32);
        ss += __shfl_xor(ss, off, 32);
    }
    float mu = s * (1.f / CCH);
    float rs = rsqrtf(ss * (1.f / CCH) - mu * mu + 1e-5f);
    _Float16* o = outh + (size_t)row * CCH;
    #pragma unroll
    for (int i = 0; i < 20; ++i) {
        int c = lane + 32 * i;
        o[c] = (_Float16)((vals[i] - mu) * rs * g[c] + bt[c]);
    }
}

// ---------------------------------------------------------------------------
extern "C" void kernel_launch(void* const* d_in, const int* in_sizes, int n_in,
                              void* d_out, int out_size, void* d_ws, size_t ws_size,
                              hipStream_t stream) {
    const float* x        = (const float*)d_in[0];
    const float* ln_g     = (const float*)d_in[1];
    const float* ln_b     = (const float*)d_in[2];
    const float* in_proj  = (const float*)d_in[3];
    const float* conv_w   = (const float*)d_in[4];
    const float* conv_b   = (const float*)d_in[5];
    const float* x_proj   = (const float*)d_in[6];
    const float* dt_w     = (const float*)d_in[7];
    const float* dt_b     = (const float*)d_in[8];
    const float* A_log    = (const float*)d_in[9];
    const float* Dw       = (const float*)d_in[10];
    const float* out_proj = (const float*)d_in[11];
    const float* proj_w   = (const float*)d_in[12];
    const float* proj_b   = (const float*)d_in[13];
    const float* skip     = (const float*)d_in[14];
    float* out = (float*)d_out;
    (void)in_sizes; (void)n_in; (void)out_size; (void)ws_size;

    char* ws = (char*)d_ws;
    size_t off = 0;
    auto take = [&](size_t bytes) -> char* {
        char* p = ws + off;
        off += (bytes + 255) & ~(size_t)255;
        return p;
    };

    const size_t RL = (size_t)NSEQ * LSEQ;                    // 163840 rows
    float*    chunks_f  = (float*)   take(RL * DMC * 4);
    _Float16* chunks_h  = (_Float16*)take(RL * DMC * 2);
    _Float16* xz_h      = (_Float16*)take(RL * 2 * DI * 2);
    float*    xc_f      = (float*)   take(RL * DI * 4);
    _Float16* xc_h      = (_Float16*)take(RL * DI * 2);       // reused as yg
    float*    dbc       = (float*)   take(RL * NDBC * 4);
    float*    delta     = (float*)   take(RL * DI * 4);
    float*    yscan     = (float*)   take(RL * DI * 4);
    float*    ybuf      = (float*)   take((size_t)BSZ * LSEQ * CCH * 4);
    _Float16* ybuf_h    = (_Float16*)take((size_t)BSZ * LSEQ * CCH * 2);
    _Float16* inproj_h  = (_Float16*)take(DMC * 2 * DI * 2);
    _Float16* xproj_h   = (_Float16*)take(DI * 64 * 2);       // padded 40 -> 64
    _Float16* outproj_h = (_Float16*)take(DI * DMC * 2);
    _Float16* proj_h    = (_Float16*)take(CCH * CCH * 2);

    // weight conversion to f16 (x_proj zero-padded to ldw = 64)
    cvt_pad_kernel<<<(DMC * 2 * DI + 255) / 256, 256, 0, stream>>>(in_proj, inproj_h, DMC, 2 * DI, 2 * DI);
    cvt_pad_kernel<<<(DI * 64 + 255) / 256, 256, 0, stream>>>(x_proj, xproj_h, DI, NDBC, 64);
    cvt_pad_kernel<<<(DI * DMC + 255) / 256, 256, 0, stream>>>(out_proj, outproj_h, DI, DMC, DMC);
    cvt_pad_kernel<<<(CCH * CCH + 255) / 256, 256, 0, stream>>>(proj_w, proj_h, CCH, CCH, CCH);

    // 1) LN over C, emit chunk layout
    ln1_kernel<<<dim3(LSEQ / 32, BSZ), 256, 0, stream>>>(x, ln_g, ln_b, chunks_f, chunks_h);

    // 2) in_proj: (163840 x 128) x (128 x 512) -> xz (f16)
    gemm_wmma_kernel<0><<<dim3((unsigned)(RL / BM), (2 * DI) / BN), 256, 0, stream>>>(
        chunks_h, inproj_h, nullptr, xz_h, (int)RL, 2 * DI, DMC, 2 * DI, nullptr, nullptr, nullptr);

    // 3) depthwise causal conv + SiLU -> xc (f32 + f16)
    conv_kernel<<<dim3(LSEQ / 64, NSEQ), 256, 0, stream>>>(xz_h, conv_w, conv_b, xc_f, xc_h);

    // 4) x_proj: (163840 x 256) x (256 x 40 padded 64) -> dbc (f32)
    gemm_wmma_kernel<1><<<dim3((unsigned)(RL / BM), 1), 256, 0, stream>>>(
        xc_h, xproj_h, dbc, nullptr, (int)RL, NDBC, DI, 64, nullptr, nullptr, nullptr);

    // 5) dt_proj + softplus -> delta
    dtproj_kernel<<<(unsigned)(RL / 16), 256, 0, stream>>>(dbc, dt_w, dt_b, delta);

    // 6) selective scan
    scan_kernel<<<NSEQ, 256, 0, stream>>>(delta, dbc, xc_f, A_log, yscan);

    // 7) gate -> yg (f16, reuses xc_h storage)
    gate_kernel<<<(unsigned)((RL * DI) / 256), 256, 0, stream>>>(yscan, xc_f, xz_h, Dw, xc_h);

    // 8) out_proj + skip, scatter to (B,L,C)
    gemm_wmma_kernel<2><<<dim3((unsigned)(RL / BM), DMC / BN), 256, 0, stream>>>(
        xc_h, outproj_h, ybuf, nullptr, (int)RL, DMC, DI, DMC, chunks_f, nullptr, skip);

    // 9) LN over C (contiguous rows)
    ln2_kernel<<<(BSZ * LSEQ) / 8, 256, 0, stream>>>(ybuf, ln_g, ln_b, ybuf_h);

    // 10) final proj + bias, transposed store to (B,C,L) = d_out
    gemm_wmma_kernel<3><<<dim3((BSZ * LSEQ) / BM, CCH / BN), 256, 0, stream>>>(
        ybuf_h, proj_h, out, nullptr, BSZ * LSEQ, CCH, CCH, CCH, nullptr, proj_b, nullptr);
}
